// PVConv_31490700215020
// MI455X (gfx1250) — matrix operand order repaired and checked by reading, not compiled, not run on hardware
//
#include <hip/hip_runtime.h>
#include <hip/hip_bf16.h>

// ---------------------------------------------------------------------------
// Problem constants (match reference)
// ---------------------------------------------------------------------------
#define BB     8
#define NP     4096
#define CIN1   64
#define COUT   128
#define RR     32
#define R3     (RR*RR*RR)
#define GROUPS 8
#define GN_EPS 1e-5f

typedef __attribute__((ext_vector_type(16))) __bf16 v16bf;
typedef __attribute__((ext_vector_type(8)))  float  v8f;

union AFrag { v16bf v; uint4 u[2]; };

__device__ __forceinline__ unsigned short f2bf(float f) {
    unsigned int u = __float_as_uint(f);
    unsigned int r = (u + 0x7FFFu + ((u >> 16) & 1u)) >> 16;
    return (unsigned short)r;
}

__device__ __forceinline__ v8f v8f_zero() {
    v8f z;
#pragma unroll
    for (int i = 0; i < 8; ++i) z[i] = 0.0f;
    return z;
}

// stage one 8 KB B-fragment tile (8 nt x 32 lanes x 16 halfs) into LDS
__device__ __forceinline__ void stageB(unsigned short* __restrict__ dst,
                                       const unsigned short* __restrict__ wrep,
                                       int tile, int tid) {
    const uint4* src = (const uint4*)(wrep + (size_t)tile * 4096);
    uint4* d = (uint4*)dst;
#pragma unroll
    for (int i = 0; i < 4; ++i) d[tid + 128 * i] = src[tid + 128 * i];
}

// ---------------------------------------------------------------------------
// 0) zero a region of workspace (uint4 granularity)
// ---------------------------------------------------------------------------
__global__ void k_zero(uint4* __restrict__ p, size_t n4) {
    size_t i = (size_t)blockIdx.x * blockDim.x + threadIdx.x;
    size_t stride = (size_t)gridDim.x * blockDim.x;
    uint4 z; z.x = z.y = z.z = z.w = 0u;
    for (; i < n4; i += stride) p[i] = z;
}

// ---------------------------------------------------------------------------
// 1) per-batch mean + scale (2*max||c-mean||)
// ---------------------------------------------------------------------------
__global__ __launch_bounds__(256) void k_scale(const float* __restrict__ coords,
                                               float* __restrict__ meanScale) {
    __shared__ float s0[256], s1[256], s2[256];
    int b = blockIdx.x, tid = threadIdx.x;
    float sx = 0.f, sy = 0.f, sz = 0.f;
    for (int n = tid; n < NP; n += 256) {
        const float* p = coords + ((size_t)b * NP + n) * 3;
        sx += p[0]; sy += p[1]; sz += p[2];
    }
    s0[tid] = sx; s1[tid] = sy; s2[tid] = sz;
    __syncthreads();
    for (int off = 128; off > 0; off >>= 1) {
        if (tid < off) { s0[tid] += s0[tid+off]; s1[tid] += s1[tid+off]; s2[tid] += s2[tid+off]; }
        __syncthreads();
    }
    float mx = s0[0] / NP, my = s1[0] / NP, mz = s2[0] / NP;
    __syncthreads();
    float mm = 0.f;
    for (int n = tid; n < NP; n += 256) {
        const float* p = coords + ((size_t)b * NP + n) * 3;
        float dx = p[0]-mx, dy = p[1]-my, dz = p[2]-mz;
        mm = fmaxf(mm, dx*dx + dy*dy + dz*dz);
    }
    s0[tid] = mm;
    __syncthreads();
    for (int off = 128; off > 0; off >>= 1) {
        if (tid < off) s0[tid] = fmaxf(s0[tid], s0[tid+off]);
        __syncthreads();
    }
    if (tid == 0) {
        float* o = meanScale + b * 4;
        o[0] = mx; o[1] = my; o[2] = mz;
        o[3] = 2.0f * sqrtf(s0[0]);
    }
}

// ---------------------------------------------------------------------------
// 2) per-point: norm coords, voxel index, count atomics
// ---------------------------------------------------------------------------
__global__ void k_coords(const float* __restrict__ coords, const float* __restrict__ meanScale,
                         float* __restrict__ ncoords, int* __restrict__ flatIdx,
                         float* __restrict__ counts) {
    int idx = blockIdx.x * blockDim.x + threadIdx.x;
    if (idx >= BB * NP) return;
    int b = idx / NP;
    const float* ms = meanScale + b * 4;
    const float* p  = coords + (size_t)idx * 3;
    float inv = 1.0f / ms[3];
    int vox[3];
    float* nc = ncoords + (size_t)idx * 3;
    for (int k = 0; k < 3; ++k) {
        float c = (p[k] - ms[k]) * inv + 0.5f;
        c = fminf(fmaxf(c * RR, 0.0f), (float)(RR - 1));
        nc[k] = c;
        vox[k] = (int)rintf(c);
    }
    int flat = (vox[0] * RR + vox[1]) * RR + vox[2];
    flatIdx[idx] = flat;
    atomicAdd(&counts[(size_t)b * R3 + flat], 1.0f);
}

// ---------------------------------------------------------------------------
// 3) scatter-add features into sums grid [B][R3][CIN1]
// ---------------------------------------------------------------------------
__global__ void k_scatter(const float* __restrict__ features, const int* __restrict__ flatIdx,
                          float* __restrict__ sums) {
    size_t idx = (size_t)blockIdx.x * blockDim.x + threadIdx.x;
    if (idx >= (size_t)BB * NP * CIN1) return;
    int c  = idx & (CIN1 - 1);
    int n  = (idx >> 6) % NP;
    int b  = (int)(idx / ((size_t)NP * CIN1));
    int fl = flatIdx[b * NP + n];
    float v = features[((size_t)b * CIN1 + c) * NP + n];
    atomicAdd(&sums[((size_t)b * R3 + fl) * CIN1 + c], v);
}

// ---------------------------------------------------------------------------
// 4) finalize grid: mean + convert to bf16, spatial-major [B][R3][CIN1]
// ---------------------------------------------------------------------------
__global__ void k_finalize(const float* __restrict__ sums, const float* __restrict__ counts,
                           unsigned short* __restrict__ gridbf) {
    size_t idx = (size_t)blockIdx.x * blockDim.x + threadIdx.x;
    if (idx >= (size_t)BB * R3 * CIN1) return;
    size_t bp = idx >> 6;  // b*R3+pos
    float cnt = counts[bp];
    float g = sums[idx] / fmaxf(cnt, 1.0f);
    gridbf[idx] = f2bf(g);
}

// ---------------------------------------------------------------------------
// 5) repack weights [COUT][CIN][T] f32 -> bf16 WMMA B-fragment tiles
//    dst[tile][nt][lane][j], tile = t*(CIN/32)+kc, K=(lane>=16?8:0)+(j&7)+(j>=8?16:0), n=lane&15
// ---------------------------------------------------------------------------
__global__ void k_repack(const float* __restrict__ w, unsigned short* __restrict__ dst,
                         int T, int CIN) {
    int idx = blockIdx.x * blockDim.x + threadIdx.x;
    int total = T * (CIN / 32) * 8 * 32 * 16;
    if (idx >= total) return;
    int j    = idx & 15;
    int l    = (idx >> 4) & 31;
    int nt   = (idx >> 9) & 7;
    int tile = idx >> 12;
    int kc = tile % (CIN / 32);
    int t  = tile / (CIN / 32);
    int Kl = ((l >> 4) << 3) + (j & 7) + ((j >> 3) << 4);       // 0..31
    int cin = kc * 32 + Kl;
    int o   = nt * 16 + (l & 15);
    float v = w[((size_t)o * CIN + cin) * T + t];
    dst[idx] = f2bf(v);
}

// ---------------------------------------------------------------------------
// 6) transpose+convert features -> bf16 point-major [B][N][CIN1]
// ---------------------------------------------------------------------------
__global__ void k_feat_t(const float* __restrict__ features, unsigned short* __restrict__ fbf) {
    size_t idx = (size_t)blockIdx.x * blockDim.x + threadIdx.x;
    if (idx >= (size_t)BB * NP * CIN1) return;
    int c = idx & (CIN1 - 1);
    int n = (idx >> 6) % NP;
    int b = (int)(idx / ((size_t)NP * CIN1));
    fbf[idx] = f2bf(features[((size_t)b * CIN1 + c) * NP + n]);
}

// ---------------------------------------------------------------------------
// 7) implicit-GEMM 3x3x3 conv via WMMA bf16 (f32 accum).
//    Block = 4 waves; each wave: 32 spatial rows x 128 couts (16 wmma tiles).
//    B-tiles double-buffered in LDS (2 x 8 KB), shared by all 4 waves.
//    GroupNorm partial stats (sum, sumsq incl. bias) fused via wave reduction.
// ---------------------------------------------------------------------------
template <int CIN_T>
__global__ __launch_bounds__(128) void k_conv(const unsigned short* __restrict__ gridbf,
                                              const unsigned short* __restrict__ wrep,
                                              const float* __restrict__ bias,
                                              float* __restrict__ outraw,
                                              float2* __restrict__ partial) {
    constexpr int KC = CIN_T / 32;
    constexpr int NK = 27 * KC;
    __shared__ __align__(32) unsigned short lw[2][8 * 32 * 16];  // 2 x 8 KB

    int blk  = blockIdx.x;                 // B * (R3/128)
    int b    = blk / (R3 / 128);
    int t128 = blk % (R3 / 128);
    int tid  = threadIdx.x;
    int wave = tid >> 5;
    int lane = tid & 31;
    int rowBase = t128 * 128 + wave * 32;

    int m = lane & 15;
    int pos0 = rowBase + m;
    int pos1 = rowBase + 16 + m;
    int z0 = pos0 & 31, y0 = (pos0 >> 5) & 31, x0 = pos0 >> 10;
    int z1 = pos1 & 31, y1 = (pos1 >> 5) & 31, x1 = pos1 >> 10;
    int kHalf = (lane >> 4) ? 8 : 0;

    v8f acc0[8], acc1[8];
#pragma unroll
    for (int i = 0; i < 8; ++i) { acc0[i] = v8f_zero(); acc1[i] = v8f_zero(); }

    const unsigned short* gbase = gridbf + (size_t)b * R3 * CIN_T;

    stageB(lw[0], wrep, 0, tid);
    __syncthreads();

    for (int t = 0; t < 27; ++t) {
        int dx = t / 9 - 1, dy = (t / 3) % 3 - 1, dz = t % 3 - 1;
        int xa = x0 + dx, ya = y0 + dy, za = z0 + dz;
        int xb = x1 + dx, yb = y1 + dy, zb = z1 + dz;
        bool in0 = (unsigned)xa < (unsigned)RR && (unsigned)ya < (unsigned)RR &&
                   (unsigned)za < (unsigned)RR;
        bool in1 = (unsigned)xb < (unsigned)RR && (unsigned)yb < (unsigned)RR &&
                   (unsigned)zb < (unsigned)RR;
        const unsigned short* arow0 = gbase + (size_t)(in0 ? ((xa*RR+ya)*RR+za) : 0) * CIN_T;
        const unsigned short* arow1 = gbase + (size_t)(in1 ? ((xb*RR+yb)*RR+zb) : 0) * CIN_T;

        for (int kc = 0; kc < KC; ++kc) {
            int tile = t * KC + kc;
            int buf  = tile & 1;
            if (tile + 1 < NK) stageB(lw[buf ^ 1], wrep, tile + 1, tid);

            uint4 zv; zv.x = zv.y = zv.z = zv.w = 0u;
            AFrag a0, a1;
            int cb = kc * 32 + kHalf;
            if (in0) { a0.u[0] = *(const uint4*)(arow0 + cb);
                       a0.u[1] = *(const uint4*)(arow0 + cb + 16); }
            else     { a0.u[0] = zv; a0.u[1] = zv; }
            if (in1) { a1.u[0] = *(const uint4*)(arow1 + cb);
                       a1.u[1] = *(const uint4*)(arow1 + cb + 16); }
            else     { a1.u[0] = zv; a1.u[1] = zv; }

#pragma unroll
            for (int nt = 0; nt < 8; ++nt) {
                v16bf bf = *(const v16bf*)(lw[buf] + (nt * 32 + lane) * 16);
                acc0[nt] = __builtin_amdgcn_wmma_f32_16x16x32_bf16(
                    false, a0.v, false, bf, (short)0, acc0[nt], false, false);
                acc1[nt] = __builtin_amdgcn_wmma_f32_16x16x32_bf16(
                    false, a1.v, false, bf, (short)0, acc1[nt], false, false);
            }
            __syncthreads();
        }
    }

    // store raw (+bias) to [B][R3][COUT] and emit GroupNorm partials
    int ncol = lane & 15;
    int mrb  = (lane >> 4) * 8;
#pragma unroll
    for (int nt = 0; nt < 8; ++nt) {
        int cout = nt * 16 + ncol;
        float bv = bias[cout];
        float s = 0.f, s2 = 0.f;
#pragma unroll
        for (int r = 0; r < 8; ++r) {
            int p0 = rowBase + mrb + r;
            int p1 = rowBase + 16 + mrb + r;
            float v0 = acc0[nt][r] + bv;
            float v1 = acc1[nt][r] + bv;
            outraw[((size_t)b * R3 + p0) * COUT + cout] = v0;
            outraw[((size_t)b * R3 + p1) * COUT + cout] = v1;
            s += v0 + v1;
            s2 += v0 * v0 + v1 * v1;
        }
#pragma unroll
        for (int off = 16; off > 0; off >>= 1) {
            s  += __shfl_xor(s,  off, 32);
            s2 += __shfl_xor(s2, off, 32);
        }
        if (lane == 0) {
            float2 p; p.x = s; p.y = s2;
            partial[((size_t)blk * 4 + wave) * 8 + nt] = p;
        }
    }
}

// ---------------------------------------------------------------------------
// 8) pointwise GEMM branch: pf = pt_w @ features  (K=64, one tap) + GN partials
// ---------------------------------------------------------------------------
__global__ __launch_bounds__(128) void k_pt_gemm(const unsigned short* __restrict__ fbf,
                                                 const unsigned short* __restrict__ wrep,
                                                 const float* __restrict__ bias,
                                                 float* __restrict__ outraw,
                                                 float2* __restrict__ partial) {
    __shared__ __align__(32) unsigned short lw[2][8 * 32 * 16];

    int blk  = blockIdx.x;                 // B * (NP/128)
    int b    = blk / (NP / 128);
    int t128 = blk % (NP / 128);
    int tid  = threadIdx.x;
    int wave = tid >> 5;
    int lane = tid & 31;
    int rowBase = t128 * 128 + wave * 32;

    int m = lane & 15;
    const unsigned short* arow0 = fbf + ((size_t)b * NP + rowBase + m) * CIN1;
    const unsigned short* arow1 = fbf + ((size_t)b * NP + rowBase + 16 + m) * CIN1;
    int kHalf = (lane >> 4) ? 8 : 0;

    v8f acc0[8], acc1[8];
#pragma unroll
    for (int i = 0; i < 8; ++i) { acc0[i] = v8f_zero(); acc1[i] = v8f_zero(); }

    stageB(lw[0], wrep, 0, tid);
    __syncthreads();

    for (int kc = 0; kc < 2; ++kc) {
        int buf = kc & 1;
        if (kc + 1 < 2) stageB(lw[buf ^ 1], wrep, kc + 1, tid);
        AFrag a0, a1;
        int cb = kc * 32 + kHalf;
        a0.u[0] = *(const uint4*)(arow0 + cb);
        a0.u[1] = *(const uint4*)(arow0 + cb + 16);
        a1.u[0] = *(const uint4*)(arow1 + cb);
        a1.u[1] = *(const uint4*)(arow1 + cb + 16);
#pragma unroll
        for (int nt = 0; nt < 8; ++nt) {
            v16bf bf = *(const v16bf*)(lw[buf] + (nt * 32 + lane) * 16);
            acc0[nt] = __builtin_amdgcn_wmma_f32_16x16x32_bf16(
                false, a0.v, false, bf, (short)0, acc0[nt], false, false);
            acc1[nt] = __builtin_amdgcn_wmma_f32_16x16x32_bf16(
                false, a1.v, false, bf, (short)0, acc1[nt], false, false);
        }
        __syncthreads();
    }

    int ncol = lane & 15;
    int mrb  = (lane >> 4) * 8;
#pragma unroll
    for (int nt = 0; nt < 8; ++nt) {
        int cout = nt * 16 + ncol;
        float bv = bias[cout];
        float s = 0.f, s2 = 0.f;
#pragma unroll
        for (int r = 0; r < 8; ++r) {
            int n0 = rowBase + mrb + r;
            int n1 = rowBase + 16 + mrb + r;
            float v0 = acc0[nt][r] + bv;
            float v1 = acc1[nt][r] + bv;
            outraw[((size_t)b * NP + n0) * COUT + cout] = v0;
            outraw[((size_t)b * NP + n1) * COUT + cout] = v1;
            s += v0 + v1;
            s2 += v0 * v0 + v1 * v1;
        }
#pragma unroll
        for (int off = 16; off > 0; off >>= 1) {
            s  += __shfl_xor(s,  off, 32);
            s2 += __shfl_xor(s2, off, 32);
        }
        if (lane == 0) {
            float2 p; p.x = s; p.y = s2;
            partial[((size_t)blk * 4 + wave) * 8 + nt] = p;
        }
    }
}

// ---------------------------------------------------------------------------
// 9) reduce per-wave GN partials -> mean/var per (b, group)
//    partial entry index: (blk*4+wave)*8 + g, blk = b*blocksPerB + ...
// ---------------------------------------------------------------------------
__global__ __launch_bounds__(256) void k_stats_reduce(const float2* __restrict__ partial,
                                                      int blocksPerB, int S,
                                                      float* __restrict__ stats) {
    __shared__ float s0[256], s1[256];
    int b = blockIdx.x / GROUPS, g = blockIdx.x % GROUPS, tid = threadIdx.x;
    int entries = blocksPerB * 4;
    float s = 0.f, s2 = 0.f;
    for (int i = tid; i < entries; i += 256) {
        float2 p = partial[((size_t)b * entries + i) * 8 + g];
        s += p.x; s2 += p.y;
    }
    s0[tid] = s; s1[tid] = s2;
    __syncthreads();
    for (int off = 128; off > 0; off >>= 1) {
        if (tid < off) { s0[tid] += s0[tid+off]; s1[tid] += s1[tid+off]; }
        __syncthreads();
    }
    if (tid == 0) {
        float cnt  = (float)S * (COUT / GROUPS);
        float mean = s0[0] / cnt;
        float var  = s1[0] / cnt - mean * mean;
        stats[(b * GROUPS + g) * 2 + 0] = mean;
        stats[(b * GROUPS + g) * 2 + 1] = var;
    }
}

// ---------------------------------------------------------------------------
// 10) GroupNorm apply + swish; optional bf16 and/or f32 outputs
// ---------------------------------------------------------------------------
__global__ void k_gn_apply(const float* __restrict__ x, const float* __restrict__ stats,
                           const float* __restrict__ gamma, const float* __restrict__ beta,
                           unsigned short* __restrict__ outbf, float* __restrict__ outf, int S) {
    size_t idx = (size_t)blockIdx.x * blockDim.x + threadIdx.x;
    size_t total = (size_t)BB * S * COUT;
    if (idx >= total) return;
    int c  = idx & (COUT - 1);
    int b  = (int)(idx / ((size_t)S * COUT));
    int g  = c >> 4;
    float mean = stats[(b * GROUPS + g) * 2 + 0];
    float var  = stats[(b * GROUPS + g) * 2 + 1];
    float v = (x[idx] - mean) * rsqrtf(var + GN_EPS) * gamma[c] + beta[c];
    float sw = v / (1.0f + __expf(-v));
    if (outbf) outbf[idx] = f2bf(sw);
    if (outf)  outf[idx]  = sw;
}

// ---------------------------------------------------------------------------
// 11) trilinear devoxelize + point-branch GN/swish + add -> out [B][COUT][N]
// ---------------------------------------------------------------------------
__global__ __launch_bounds__(128) void k_devox(const float* __restrict__ h2,
                                               const float* __restrict__ pfraw,
                                               const float* __restrict__ stats,
                                               const float* __restrict__ gamma,
                                               const float* __restrict__ beta,
                                               const float* __restrict__ ncoords,
                                               float* __restrict__ out) {
    int b = blockIdx.x / NP, n = blockIdx.x % NP, c = threadIdx.x;
    const float* p = ncoords + ((size_t)b * NP + n) * 3;
    float nx = p[0], ny = p[1], nz = p[2];
    float fx = floorf(nx), fy = floorf(ny), fz = floorf(nz);
    float dx = nx - fx, dy = ny - fy, dz = nz - fz;
    int x0 = (int)fx, y0 = (int)fy, z0 = (int)fz;
    int x1 = min(x0 + 1, RR - 1), y1 = min(y0 + 1, RR - 1), z1 = min(z0 + 1, RR - 1);

    float acc = 0.f;
#pragma unroll
    for (int ix = 0; ix < 2; ++ix)
#pragma unroll
        for (int iy = 0; iy < 2; ++iy)
#pragma unroll
            for (int iz = 0; iz < 2; ++iz) {
                int xi = ix ? x1 : x0, yi = iy ? y1 : y0, zi = iz ? z1 : z0;
                float w = (ix ? dx : 1.f - dx) * (iy ? dy : 1.f - dy) * (iz ? dz : 1.f - dz);
                size_t gi = ((size_t)b * R3 + ((xi * RR + yi) * RR + zi)) * COUT + c;
                acc += h2[gi] * w;
            }

    float v = pfraw[((size_t)b * NP + n) * COUT + c];
    int g = c >> 4;
    float mean = stats[(b * GROUPS + g) * 2 + 0];
    float var  = stats[(b * GROUPS + g) * 2 + 1];
    v = (v - mean) * rsqrtf(var + GN_EPS) * gamma[c] + beta[c];
    float sw = v / (1.0f + __expf(-v));

    out[((size_t)b * COUT + c) * NP + n] = acc + sw;
}

// ---------------------------------------------------------------------------
// host-side launch
// ---------------------------------------------------------------------------
extern "C" void kernel_launch(void* const* d_in, const int* in_sizes, int n_in,
                              void* d_out, int out_size, void* d_ws, size_t ws_size,
                              hipStream_t stream) {
    const float* features = (const float*)d_in[0];
    const float* coords   = (const float*)d_in[1];
    const float* conv1_w  = (const float*)d_in[2];
    const float* conv1_b  = (const float*)d_in[3];
    const float* gn1_g    = (const float*)d_in[4];
    const float* gn1_b    = (const float*)d_in[5];
    const float* conv2_w  = (const float*)d_in[6];
    const float* conv2_b  = (const float*)d_in[7];
    const float* gn2_g    = (const float*)d_in[8];
    const float* gn2_b    = (const float*)d_in[9];
    const float* pt_w     = (const float*)d_in[10];
    const float* pt_b     = (const float*)d_in[11];
    const float* ptgn_g   = (const float*)d_in[12];
    const float* ptgn_b   = (const float*)d_in[13];
    float* out = (float*)d_out;

    char* ws = (char*)d_ws;
    size_t off = 0;
    auto alloc = [&](size_t bytes) { size_t o = off; off += (bytes + 255) & ~(size_t)255; return o; };

    size_t oMean   = alloc((size_t)BB * 4 * 4);
    size_t oNC     = alloc((size_t)BB * NP * 3 * 4);
    size_t oFlat   = alloc((size_t)BB * NP * 4);
    size_t oCnt    = alloc((size_t)BB * R3 * 4);
    size_t oSum    = alloc((size_t)BB * R3 * CIN1 * 4);
    size_t oGridBF = alloc((size_t)BB * R3 * CIN1 * 2);
    size_t oRaw    = alloc((size_t)BB * R3 * COUT * 4);   // conv1 raw, reused for conv2 raw
    size_t oAct1   = alloc((size_t)BB * R3 * COUT * 2);   // bf16 conv2 input
    size_t oH2     = alloc((size_t)BB * R3 * COUT * 4);   // f32 post-GN2 grid
    size_t oW1     = alloc((size_t)27 * (CIN1/32) * 8 * 512 * 2);
    size_t oW2     = alloc((size_t)27 * (COUT/32) * 8 * 512 * 2);
    size_t oWPt    = alloc((size_t)1 * (CIN1/32) * 8 * 512 * 2);
    size_t oFeatBF = alloc((size_t)BB * NP * CIN1 * 2);
    size_t oPF     = alloc((size_t)BB * NP * COUT * 4);
    size_t oPart   = alloc((size_t)BB * (R3/128) * 4 * 8 * 8);  // float2 per (blk,wave,nt)
    size_t oStats  = alloc((size_t)3 * BB * GROUPS * 2 * 4);
    (void)ws_size; (void)n_in; (void)in_sizes; (void)out_size;

    float* meanScale = (float*)(ws + oMean);
    float* ncoords   = (float*)(ws + oNC);
    int*   flatIdx   = (int*)(ws + oFlat);
    float* counts    = (float*)(ws + oCnt);
    float* sums      = (float*)(ws + oSum);
    unsigned short* gridbf = (unsigned short*)(ws + oGridBF);
    float* raw       = (float*)(ws + oRaw);
    unsigned short* act1 = (unsigned short*)(ws + oAct1);
    float* h2        = (float*)(ws + oH2);
    unsigned short* w1 = (unsigned short*)(ws + oW1);
    unsigned short* w2 = (unsigned short*)(ws + oW2);
    unsigned short* wpt = (unsigned short*)(ws + oWPt);
    unsigned short* featbf = (unsigned short*)(ws + oFeatBF);
    float* pfraw     = (float*)(ws + oPF);
    float2* partial  = (float2*)(ws + oPart);
    float* stats1    = (float*)(ws + oStats);
    float* stats2    = stats1 + BB * GROUPS * 2;
    float* stats3    = stats2 + BB * GROUPS * 2;

    // zero counts + sums (contiguous, both 256-aligned and adjacent)
    {
        size_t zbytes = (size_t)BB * R3 * 4 + (size_t)BB * R3 * CIN1 * 4;
        k_zero<<<4096, 256, 0, stream>>>((uint4*)(ws + oCnt), zbytes / 16);
    }

    // voxelize
    k_scale<<<BB, 256, 0, stream>>>(coords, meanScale);
    k_coords<<<(BB * NP + 255) / 256, 256, 0, stream>>>(coords, meanScale, ncoords, flatIdx, counts);
    k_scatter<<<(int)(((size_t)BB * NP * CIN1 + 255) / 256), 256, 0, stream>>>(features, flatIdx, sums);
    k_finalize<<<(int)(((size_t)BB * R3 * CIN1 + 255) / 256), 256, 0, stream>>>(sums, counts, gridbf);

    // weight repack (bf16 fragment tiles)
    k_repack<<<(27 * (CIN1/32) * 4096 + 255) / 256, 256, 0, stream>>>(conv1_w, w1, 27, CIN1);
    k_repack<<<(27 * (COUT/32) * 4096 + 255) / 256, 256, 0, stream>>>(conv2_w, w2, 27, COUT);
    k_repack<<<(1 * (CIN1/32) * 4096 + 255) / 256, 256, 0, stream>>>(pt_w, wpt, 1, CIN1);
    k_feat_t<<<(int)(((size_t)BB * NP * CIN1 + 255) / 256), 256, 0, stream>>>(features, featbf);

    // conv1 -> GN1 -> swish -> bf16
    k_conv<CIN1><<<BB * (R3 / 128), 128, 0, stream>>>(gridbf, w1, conv1_b, raw, partial);
    k_stats_reduce<<<BB * GROUPS, 256, 0, stream>>>(partial, R3 / 128, R3, stats1);
    k_gn_apply<<<(int)(((size_t)BB * R3 * COUT + 255) / 256), 256, 0, stream>>>(
        raw, stats1, gn1_g, gn1_b, act1, (float*)nullptr, R3);

    // conv2 -> GN2 -> swish -> f32
    k_conv<COUT><<<BB * (R3 / 128), 128, 0, stream>>>(act1, w2, conv2_b, raw, partial);
    k_stats_reduce<<<BB * GROUPS, 256, 0, stream>>>(partial, R3 / 128, R3, stats2);
    k_gn_apply<<<(int)(((size_t)BB * R3 * COUT + 255) / 256), 256, 0, stream>>>(
        raw, stats2, gn2_g, gn2_b, (unsigned short*)nullptr, h2, R3);

    // point branch GEMM -> GN stats
    k_pt_gemm<<<BB * (NP / 128), 128, 0, stream>>>(featbf, wpt, pt_b, pfraw, partial);
    k_stats_reduce<<<BB * GROUPS, 256, 0, stream>>>(partial, NP / 128, NP, stats3);

    // devoxelize + point GN/swish + add
    k_devox<<<BB * NP, 128, 0, stream>>>(h2, pfraw, stats3, ptgn_g, ptgn_b, ncoords, out);
}
